// MultiheadLatentAttention_6004364280519
// MI455X (gfx1250) — matrix operand order repaired and checked by reading, not compiled, not run on hardware
//
#include <hip/hip_runtime.h>
#include <hip/hip_bf16.h>

// ---------------------------------------------------------------------------
// MLA forward for gfx1250 (CDNA5). All heavy math via v_wmma_f32_16x16x32_bf16.
// GEMM tile staging via Tensor Data Mover (tensor_load_to_lds) + double buffer.
// ---------------------------------------------------------------------------

typedef __bf16 bf16_t;
typedef __attribute__((ext_vector_type(16))) __bf16 v16bf;
typedef __attribute__((ext_vector_type(8)))  __bf16 v8bf;
typedef __attribute__((ext_vector_type(8)))  float  v8f;
typedef __attribute__((ext_vector_type(4)))  unsigned int u32x4;
typedef __attribute__((ext_vector_type(8)))  int i32x8;
typedef __attribute__((ext_vector_type(4)))  int i32x4;

#define S_LEN 1024
#define HIDN  7168
#define NH    128
#define DQC   1536
#define DKVC  512
#define DHD   128
#define DRR   64
#define DQK   192   // DHD + DRR

#if defined(__gfx1250__) && __has_builtin(__builtin_amdgcn_tensor_load_to_lds)
#define HAS_TDM 1
#else
#define HAS_TDM 0
#endif

// ---------------------------------------------------------------------------
// fp32 -> bf16 conversion (vectorized 8/thread)
// ---------------------------------------------------------------------------
__global__ __launch_bounds__(256)
void cvt_f32_bf16(const float* __restrict__ src, bf16_t* __restrict__ dst, long n) {
    long base = ((long)blockIdx.x * 256 + threadIdx.x) * 8;
    if (base + 7 < n) {
        float4 a = *(const float4*)(src + base);
        float4 b = *(const float4*)(src + base + 4);
        v8bf o;
        o[0] = (bf16_t)a.x; o[1] = (bf16_t)a.y; o[2] = (bf16_t)a.z; o[3] = (bf16_t)a.w;
        o[4] = (bf16_t)b.x; o[5] = (bf16_t)b.y; o[6] = (bf16_t)b.z; o[7] = (bf16_t)b.w;
        *(v8bf*)(dst + base) = o;
    } else {
        for (int i = 0; i < 8; ++i) {
            long idx = base + i;
            if (idx < n) dst[idx] = (bf16_t)src[idx];
        }
    }
}

#if HAS_TDM
// ---------------------------------------------------------------------------
// Issue one TDM 2-D tile load: tile_d0 x tile_d1 elements (2-byte) from
// row-major tensor with row stride stride0 (elements) into LDS (contiguous
// tile_d0-element rows). OOB rows/cols (beyond tensor_d0/d1) are zero-filled
// by the TDM. D# packing per CDNA5 ISA ch.8 (group0 + group1, 2-D form).
// This toolchain exposes the 6-arg builtin:
//   (u32x4 g0, i32x8 g1, i32x4 g2, i32x4 g3, i32x8 extra, imm cpol)
// ---------------------------------------------------------------------------
__device__ __forceinline__ void tdm_load_2d(const bf16_t* gsrc, unsigned lds_byte_addr,
                                            unsigned tensor_d0, unsigned tensor_d1,
                                            long stride0, unsigned tile_d0, unsigned tile_d1)
{
    unsigned long long ga = (unsigned long long)(size_t)gsrc;
    u32x4 g0;
    g0[0] = 1u;                                            // count=1, user mode
    g0[1] = lds_byte_addr;                                 // lds_addr
    g0[2] = (unsigned)(ga & 0xFFFFFFFFull);                // global_addr[31:0]
    g0[3] = (unsigned)((ga >> 32) & 0x1FFFFFFull)          // global_addr[56:32]
          | (2u << 30);                                    // type = 2 ("image")
    i32x8 g1;
    g1[0] = (int)(1u << 16);                               // data_size=1 (2 bytes), mask=0
    g1[1] = (int)((tensor_d0 & 0xFFFFu) << 16);            // tensor_dim0[15:0] @ bits 63:48
    g1[2] = (int)((tensor_d0 >> 16) | ((tensor_d1 & 0xFFFFu) << 16));
    g1[3] = (int)((tensor_d1 >> 16) | (tile_d0 << 16));    // tile_dim0 @ bits 127:112
    g1[4] = (int)(tile_d1 & 0xFFFFu);                      // tile_dim1, tile_dim2=0
    g1[5] = (int)(unsigned)(stride0 & 0xFFFFFFFFl);        // tensor_dim0_stride[31:0]
    g1[6] = (int)(unsigned)((stride0 >> 32) & 0xFFFFl);    // stride0[47:32], stride1[15:0]=0
    g1[7] = 0;                                             // stride1[47:16]=0
    i32x4 z4 = {0, 0, 0, 0};
    i32x8 z8 = {0, 0, 0, 0, 0, 0, 0, 0};
    __builtin_amdgcn_tensor_load_to_lds(g0, g1, z4, z4, z8, 0);
}
#endif

// ---------------------------------------------------------------------------
// Generic bf16 WMMA GEMM:  C[M,N] = A[M,K] * W[N,K]^T + bias[N]
// Block tile 128x128, BK=32, 8 waves arranged 4(M) x 2(N), wave tile 32x64.
// A-fragment (16x32 bf16):  lane l holds M=l%16, K=(l/16)*8+[0..7] and 16+(l/16)*8+[0..7]
// B-fragment (32x16 bf16):  lane l holds N=l%16, K=(l/16)*8+[0..7] and 16+(l/16)*8+[0..7]
// Staging: TDM double-buffered (wave 0 issues, TENSORcnt-gated) when available.
// ---------------------------------------------------------------------------
__global__ __launch_bounds__(256)
void gemm_bf16_wmma(const bf16_t* __restrict__ A, const bf16_t* __restrict__ W,
                    const float* __restrict__ bias,
                    float* __restrict__ Cf, bf16_t* __restrict__ Cb,
                    int M, int N, int K)
{
#if HAS_TDM
    __shared__ bf16_t sA[2][128 * 32];
    __shared__ bf16_t sB[2][128 * 32];
#else
    __shared__ bf16_t sA[1][128 * 32];
    __shared__ bf16_t sB[1][128 * 32];
#endif

    const int tid   = threadIdx.x;
    const int lane  = tid & 31;
    const int wave  = tid >> 5;
    const int wm    = wave >> 1;            // 0..3
    const int wn    = wave & 1;             // 0..1
    const int mblk  = blockIdx.y * 128;
    const int nblk  = blockIdx.x * 128;
    const int lmod  = lane & 15;
    const int lhalf = lane >> 4;

    v8f acc[2][4];
    #pragma unroll
    for (int i = 0; i < 2; ++i)
        #pragma unroll
        for (int j = 0; j < 4; ++j)
            acc[i][j] = v8f{0.f,0.f,0.f,0.f,0.f,0.f,0.f,0.f};

#if HAS_TDM
    // prologue: fetch tile 0 into buffer 0
    if (wave == 0) {
        tdm_load_2d(A + (long)mblk * K, (unsigned)(size_t)&sA[0][0],
                    (unsigned)K, (unsigned)(M - mblk), (long)K, 32u, 128u);
        tdm_load_2d(W + (long)nblk * K, (unsigned)(size_t)&sB[0][0],
                    (unsigned)K, (unsigned)(N - nblk), (long)K, 32u, 128u);
        __builtin_amdgcn_s_wait_tensorcnt(0);
    }
    __syncthreads();
    int cur = 0;
#else
    const int ldr = tid >> 1;               // tile row 0..127
    const int ldc = (tid & 1) * 16;         // tile col 0 or 16
#endif

    for (int kb = 0; kb < K; kb += 32) {
#if HAS_TDM
        // async prefetch of next tile into the other buffer (overlaps WMMAs)
        const int nxt = cur ^ 1;
        if (wave == 0 && kb + 32 < K) {
            tdm_load_2d(A + (long)mblk * K + kb + 32, (unsigned)(size_t)&sA[nxt][0],
                        (unsigned)K, (unsigned)(M - mblk), (long)K, 32u, 128u);
            tdm_load_2d(W + (long)nblk * K + kb + 32, (unsigned)(size_t)&sB[nxt][0],
                        (unsigned)K, (unsigned)(N - nblk), (long)K, 32u, 128u);
        }
        const bf16_t* curA = &sA[cur][0];
        const bf16_t* curB = &sB[cur][0];
#else
        // synchronous staging fallback
        {
            int arow = mblk + ldr;
            uint4 av = make_uint4(0u, 0u, 0u, 0u);
            if (arow < M) av = *(const uint4*)(A + (long)arow * K + kb + ldc);
            *(uint4*)(&sA[0][0] + ldr * 32 + ldc) = av;
            int brow = nblk + ldr;
            uint4 bv = make_uint4(0u, 0u, 0u, 0u);
            if (brow < N) bv = *(const uint4*)(W + (long)brow * K + kb + ldc);
            *(uint4*)(&sB[0][0] + ldr * 32 + ldc) = bv;
        }
        __syncthreads();
        const bf16_t* curA = &sA[0][0];
        const bf16_t* curB = &sB[0][0];
#endif

        // A fragments (2 M-subtiles)
        v16bf afrag[2];
        #pragma unroll
        for (int ti = 0; ti < 2; ++ti) {
            int row = wm * 32 + ti * 16 + lmod;
            v8bf lo = *(const v8bf*)(curA + row * 32 + lhalf * 8);
            v8bf hi = *(const v8bf*)(curA + row * 32 + 16 + lhalf * 8);
            #pragma unroll
            for (int e = 0; e < 8; ++e) { afrag[ti][e] = lo[e]; afrag[ti][8 + e] = hi[e]; }
        }

        // B fragments (4 N-subtiles), fused with WMMAs
        #pragma unroll
        for (int tj = 0; tj < 4; ++tj) {
            int nrow = wn * 64 + tj * 16 + lmod;
            v8bf lo = *(const v8bf*)(curB + nrow * 32 + lhalf * 8);
            v8bf hi = *(const v8bf*)(curB + nrow * 32 + 16 + lhalf * 8);
            v16bf bfrag;
            #pragma unroll
            for (int e = 0; e < 8; ++e) { bfrag[e] = lo[e]; bfrag[8 + e] = hi[e]; }
            #pragma unroll
            for (int ti = 0; ti < 2; ++ti) {
                acc[ti][tj] = __builtin_amdgcn_wmma_f32_16x16x32_bf16(
                    false, afrag[ti], false, bfrag, (short)0, acc[ti][tj], false, false);
            }
        }

#if HAS_TDM
        if (wave == 0) __builtin_amdgcn_s_wait_tensorcnt(0);
        __syncthreads();
        cur = nxt;
#else
        __syncthreads();
#endif
    }

    // store: C/D layout -> lane holds rows (l/16)*8 + r, col l%16 (per 16x16 subtile)
    #pragma unroll
    for (int ti = 0; ti < 2; ++ti) {
        #pragma unroll
        for (int tj = 0; tj < 4; ++tj) {
            int n = nblk + wn * 64 + tj * 16 + lmod;
            if (n >= N) continue;
            float bv = bias ? bias[n] : 0.0f;
            #pragma unroll
            for (int r = 0; r < 8; ++r) {
                int m = mblk + wm * 32 + ti * 16 + lhalf * 8 + r;
                if (m >= M) continue;
                float c = acc[ti][tj][r] + bv;
                long idx = (long)m * N + n;
                if (Cf) Cf[idx] = c;
                if (Cb) Cb[idx] = (bf16_t)c;
            }
        }
    }
}

// ---------------------------------------------------------------------------
// RoPE helper:  j in [0,64): inv_freq = 10000^(-(j%32)/32)
// ---------------------------------------------------------------------------
__device__ __forceinline__ float rope_val(const float* __restrict__ base, int j, int s) {
    int jj = j & 31;
    float invf = __expf(-0.28782313662425574f * (float)jj);  // ln(10000)/32
    float ang  = (float)s * invf;
    float c  = __cosf(ang);
    float sn = __sinf(ang);
    float x = base[j];
    float other = (j < 32) ? -base[j + 32] : base[j - 32];
    return x * c + other * sn;
}

// Q pack: Qp[h][s][0:128] = q_c, Qp[h][s][128:192] = rope(q_r)
__global__ __launch_bounds__(256)
void pack_q(const float* __restrict__ qc, const float* __restrict__ qr, bf16_t* __restrict__ Qp) {
    long idx = (long)blockIdx.x * 256 + threadIdx.x;
    if (idx >= (long)NH * S_LEN * DQK) return;
    int d = (int)(idx % DQK);
    int s = (int)((idx / DQK) % S_LEN);
    int h = (int)(idx / ((long)DQK * S_LEN));
    float val;
    if (d < DHD) val = qc[(long)s * (NH * DHD) + h * DHD + d];
    else         val = rope_val(qr + (long)s * (NH * DRR) + h * DRR, d - DHD, s);
    Qp[idx] = (bf16_t)val;
}

// K pack: Kp[h][s][0:128] = k_c (from kv), Kp[h][s][128:192] = rope(k_r) (broadcast over h)
__global__ __launch_bounds__(256)
void pack_k(const float* __restrict__ kvf, const float* __restrict__ krf, bf16_t* __restrict__ Kp) {
    long idx = (long)blockIdx.x * 256 + threadIdx.x;
    if (idx >= (long)NH * S_LEN * DQK) return;
    int d = (int)(idx % DQK);
    int s = (int)((idx / DQK) % S_LEN);
    int h = (int)(idx / ((long)DQK * S_LEN));
    float val;
    if (d < DHD) val = kvf[(long)s * (NH * 2 * DHD) + h * (2 * DHD) + d];
    else         val = rope_val(krf + (long)s * DRR, d - DHD, s);
    Kp[idx] = (bf16_t)val;
}

// V pack transposed: Vt[h][d][s] = kv[s][h*256 + 128 + d]
__global__ __launch_bounds__(256)
void pack_vt(const float* __restrict__ kvf, bf16_t* __restrict__ Vt) {
    long idx = (long)blockIdx.x * 256 + threadIdx.x;
    if (idx >= (long)NH * DHD * S_LEN) return;
    int s = (int)(idx % S_LEN);
    int d = (int)((idx / S_LEN) % DHD);
    int h = (int)(idx / ((long)S_LEN * DHD));
    Vt[idx] = (bf16_t)kvf[(long)s * (NH * 2 * DHD) + h * (2 * DHD) + DHD + d];
}

// ---------------------------------------------------------------------------
// Flash attention: 4 waves/block, each wave owns 16 queries of one head.
// Inner loop processes 32 keys: scores via 6 WMMAs per 16-key subtile,
// online softmax (shfl_xor row reductions within the 16-lane C-layout group),
// P re-laid-out through LDS into A-fragment form, P@V via 8 WMMAs.
// ---------------------------------------------------------------------------
__global__ __launch_bounds__(128)
void mla_attention(const bf16_t* __restrict__ Qp, const bf16_t* __restrict__ Kp,
                   const bf16_t* __restrict__ Vt, bf16_t* __restrict__ Obf)
{
    __shared__ bf16_t sP[4][16 * 32];

    const int tid   = threadIdx.x;
    const int lane  = tid & 31;
    const int wave  = tid >> 5;
    const int h     = blockIdx.y;
    const int qbase = blockIdx.x * 64 + wave * 16;
    const int lmod  = lane & 15;
    const int lhalf = lane >> 4;
    const float scale = 0.07216878364870323f;   // 1/sqrt(192)

    const bf16_t* Qh = Qp + (long)h * S_LEN * DQK;
    const bf16_t* Kh = Kp + (long)h * S_LEN * DQK;
    const bf16_t* Vh = Vt + (long)h * DHD * S_LEN;

    // resident Q fragments: 6 chunks of K-depth 32
    v16bf qa[6];
    #pragma unroll
    for (int c = 0; c < 6; ++c) {
        const bf16_t* qptr = Qh + (long)(qbase + lmod) * DQK + c * 32;
        v8bf lo = *(const v8bf*)(qptr + lhalf * 8);
        v8bf hi = *(const v8bf*)(qptr + 16 + lhalf * 8);
        #pragma unroll
        for (int e = 0; e < 8; ++e) { qa[c][e] = lo[e]; qa[c][8 + e] = hi[e]; }
    }

    v8f oacc[8];
    #pragma unroll
    for (int i = 0; i < 8; ++i) oacc[i] = v8f{0.f,0.f,0.f,0.f,0.f,0.f,0.f,0.f};
    float mrow[8], lrow[8];
    #pragma unroll
    for (int r = 0; r < 8; ++r) { mrow[r] = -1e30f; lrow[r] = 0.0f; }

    for (int kb = 0; kb < S_LEN; kb += 32) {
        // ---- scores for two 16-key subtiles ----
        v8f sc[2];
        #pragma unroll
        for (int t = 0; t < 2; ++t) {
            sc[t] = v8f{0.f,0.f,0.f,0.f,0.f,0.f,0.f,0.f};
            #pragma unroll
            for (int c = 0; c < 6; ++c) {
                const bf16_t* kptr = Kh + (long)(kb + t * 16 + lmod) * DQK + c * 32;
                v8bf lo = *(const v8bf*)(kptr + lhalf * 8);
                v8bf hi = *(const v8bf*)(kptr + 16 + lhalf * 8);
                v16bf kf;
                #pragma unroll
                for (int e = 0; e < 8; ++e) { kf[e] = lo[e]; kf[8 + e] = hi[e]; }
                sc[t] = __builtin_amdgcn_wmma_f32_16x16x32_bf16(
                    false, qa[c], false, kf, (short)0, sc[t], false, false);
            }
        }

        // ---- online softmax update ----
        #pragma unroll
        for (int r = 0; r < 8; ++r) {
            float s0 = sc[0][r] * scale;
            float s1 = sc[1][r] * scale;
            float tmax = fmaxf(s0, s1);
            #pragma unroll
            for (int off = 8; off >= 1; off >>= 1)
                tmax = fmaxf(tmax, __shfl_xor(tmax, off, 32));
            float newm  = fmaxf(mrow[r], tmax);
            float alpha = __expf(mrow[r] - newm);
            float p0 = __expf(s0 - newm);
            float p1 = __expf(s1 - newm);
            float psum = p0 + p1;
            #pragma unroll
            for (int off = 8; off >= 1; off >>= 1)
                psum += __shfl_xor(psum, off, 32);
            lrow[r] = lrow[r] * alpha + psum;
            mrow[r] = newm;
            #pragma unroll
            for (int dt = 0; dt < 8; ++dt) oacc[dt][r] *= alpha;
            int prow = lhalf * 8 + r;
            sP[wave][prow * 32 + lmod]      = (bf16_t)p0;
            sP[wave][prow * 32 + 16 + lmod] = (bf16_t)p1;
        }

        // ---- reload P in A-fragment layout (16x32) ----
        v16bf pa;
        {
            const bf16_t* pptr = &sP[wave][lmod * 32];
            v8bf lo = *(const v8bf*)(pptr + lhalf * 8);
            v8bf hi = *(const v8bf*)(pptr + 16 + lhalf * 8);
            #pragma unroll
            for (int e = 0; e < 8; ++e) { pa[e] = lo[e]; pa[8 + e] = hi[e]; }
        }

        // ---- P @ V : 8 output d-subtiles ----
        #pragma unroll
        for (int dt = 0; dt < 8; ++dt) {
            const bf16_t* vptr = Vh + (long)(dt * 16 + lmod) * S_LEN + kb;
            v8bf lo = *(const v8bf*)(vptr + lhalf * 8);
            v8bf hi = *(const v8bf*)(vptr + 16 + lhalf * 8);
            v16bf vf;
            #pragma unroll
            for (int e = 0; e < 8; ++e) { vf[e] = lo[e]; vf[8 + e] = hi[e]; }
            oacc[dt] = __builtin_amdgcn_wmma_f32_16x16x32_bf16(
                false, pa, false, vf, (short)0, oacc[dt], false, false);
        }
    }

    // ---- epilogue: normalize and store o[s][h*128 + d] ----
    #pragma unroll
    for (int r = 0; r < 8; ++r) {
        float linv = 1.0f / lrow[r];
        int row = qbase + lhalf * 8 + r;
        #pragma unroll
        for (int dt = 0; dt < 8; ++dt) {
            Obf[(long)row * (NH * DHD) + h * DHD + dt * 16 + lmod] =
                (bf16_t)(oacc[dt][r] * linv);
        }
    }
}

// ---------------------------------------------------------------------------
// Host-side orchestration
// ---------------------------------------------------------------------------
extern "C" void kernel_launch(void* const* d_in, const int* in_sizes, int n_in,
                              void* d_out, int out_size, void* d_ws, size_t ws_size,
                              hipStream_t stream)
{
    const float* hin  = (const float*)d_in[0];
    const float* Wdq  = (const float*)d_in[1];
    const float* bdq  = (const float*)d_in[2];
    const float* Wdkv = (const float*)d_in[3];
    const float* bdkv = (const float*)d_in[4];
    const float* Wuq  = (const float*)d_in[5];
    const float* buq  = (const float*)d_in[6];
    const float* Wukv = (const float*)d_in[7];
    const float* bukv = (const float*)d_in[8];
    const float* Wrq  = (const float*)d_in[9];
    const float* brq  = (const float*)d_in[10];
    const float* Wrk  = (const float*)d_in[11];
    const float* brk  = (const float*)d_in[12];
    const float* Wo   = (const float*)d_in[13];
    const float* bo   = (const float*)d_in[14];

    char* ws = (char*)d_ws;
    size_t off = 0;
    auto alloc = [&](size_t bytes) -> void* {
        void* p = ws + off;
        off = (off + bytes + 255) & ~(size_t)255;
        return p;
    };

    // bf16 copies
    bf16_t* h_bf    = (bf16_t*)alloc((size_t)S_LEN * HIDN * 2);
    bf16_t* Wdq_bf  = (bf16_t*)alloc((size_t)DQC  * HIDN * 2);
    bf16_t* Wdkv_bf = (bf16_t*)alloc((size_t)DKVC * HIDN * 2);
    bf16_t* Wuq_bf  = (bf16_t*)alloc((size_t)NH * DHD * DQC * 2);
    bf16_t* Wukv_bf = (bf16_t*)alloc((size_t)NH * 2 * DHD * DKVC * 2);
    bf16_t* Wrq_bf  = (bf16_t*)alloc((size_t)NH * DRR * DQC * 2);
    bf16_t* Wrk_bf  = (bf16_t*)alloc((size_t)DRR * HIDN * 2);
    bf16_t* Wo_bf   = (bf16_t*)alloc((size_t)HIDN * NH * DHD * 2);
    // intermediates
    bf16_t* cq_bf   = (bf16_t*)alloc((size_t)S_LEN * DQC * 2);
    bf16_t* ckv_bf  = (bf16_t*)alloc((size_t)S_LEN * DKVC * 2);
    float*  qc_f    = (float*) alloc((size_t)S_LEN * NH * DHD * 4);
    float*  kv_f    = (float*) alloc((size_t)S_LEN * NH * 2 * DHD * 4);
    float*  qr_f    = (float*) alloc((size_t)S_LEN * NH * DRR * 4);
    float*  kr_f    = (float*) alloc((size_t)S_LEN * DRR * 4);
    bf16_t* Qpack   = (bf16_t*)alloc((size_t)NH * S_LEN * DQK * 2);
    bf16_t* Kpack   = (bf16_t*)alloc((size_t)NH * S_LEN * DQK * 2);
    bf16_t* Vtpack  = (bf16_t*)alloc((size_t)NH * DHD * S_LEN * 2);
    bf16_t* o_bf    = (bf16_t*)alloc((size_t)S_LEN * NH * DHD * 2);

    auto cvt = [&](const float* src, bf16_t* dst, long n) {
        int blocks = (int)((n + 2047) / 2048);
        cvt_f32_bf16<<<dim3(blocks), dim3(256), 0, stream>>>(src, dst, n);
    };
    auto gemm = [&](const bf16_t* A, const bf16_t* W, const float* bias,
                    float* Cf, bf16_t* Cb, int M, int N, int K) {
        dim3 grid((N + 127) / 128, (M + 127) / 128);
        gemm_bf16_wmma<<<grid, dim3(256), 0, stream>>>(A, W, bias, Cf, Cb, M, N, K);
    };

    // 1) convert activations + weights to bf16
    cvt(hin,  h_bf,    (long)S_LEN * HIDN);
    cvt(Wdq,  Wdq_bf,  (long)DQC  * HIDN);
    cvt(Wdkv, Wdkv_bf, (long)DKVC * HIDN);
    cvt(Wuq,  Wuq_bf,  (long)NH * DHD * DQC);
    cvt(Wukv, Wukv_bf, (long)NH * 2 * DHD * DKVC);
    cvt(Wrq,  Wrq_bf,  (long)NH * DRR * DQC);
    cvt(Wrk,  Wrk_bf,  (long)DRR * HIDN);
    cvt(Wo,   Wo_bf,   (long)HIDN * NH * DHD);

    // 2) projections
    gemm(h_bf,   Wdq_bf,  bdq,  nullptr, cq_bf,  S_LEN, DQC,          HIDN);
    gemm(h_bf,   Wdkv_bf, bdkv, nullptr, ckv_bf, S_LEN, DKVC,         HIDN);
    gemm(cq_bf,  Wuq_bf,  buq,  qc_f,    nullptr, S_LEN, NH * DHD,     DQC);
    gemm(ckv_bf, Wukv_bf, bukv, kv_f,    nullptr, S_LEN, NH * 2 * DHD, DKVC);
    gemm(cq_bf,  Wrq_bf,  brq,  qr_f,    nullptr, S_LEN, NH * DRR,     DQC);
    gemm(h_bf,   Wrk_bf,  brk,  kr_f,    nullptr, S_LEN, DRR,          HIDN);

    // 3) pack Q/K (RoPE fused) and transposed V
    {
        long nq = (long)NH * S_LEN * DQK;
        pack_q<<<dim3((int)((nq + 255) / 256)), dim3(256), 0, stream>>>(qc_f, qr_f, Qpack);
        pack_k<<<dim3((int)((nq + 255) / 256)), dim3(256), 0, stream>>>(kv_f, kr_f, Kpack);
        long nv = (long)NH * DHD * S_LEN;
        pack_vt<<<dim3((int)((nv + 255) / 256)), dim3(256), 0, stream>>>(kv_f, Vtpack);
    }

    // 4) attention: grid = (S/64 query tiles, heads), 4 waves/block
    mla_attention<<<dim3(S_LEN / 64, NH), dim3(128), 0, stream>>>(Qpack, Kpack, Vtpack, o_bf);

    // 5) output projection -> d_out (fp32)
    gemm(o_bf, Wo_bf, bo, (float*)d_out, nullptr, S_LEN, HIDN, NH * DHD);
}
